// StandardLSTM_46162308498108
// MI455X (gfx1250) — matrix operand order repaired and checked by reading, not compiled, not run on hardware
//
#include <hip/hip_runtime.h>
#include <hip/hip_bf16.h>
#include <stdint.h>

#define H 64
#define T 512
#define BATCH 1024
#define TILE 16
#define NTILES (BATCH / TILE)   // 64 batch tiles
#define GSTRIDE 273             // padded LDS stride (shift 17 banks/row -> conflict-free gate reads)

typedef __attribute__((ext_vector_type(16))) __bf16 v16bf;
typedef __attribute__((ext_vector_type(8)))  float  v8f;

union Frag16 { v16bf v; uint32_t u[8]; };
union AccF   { v8f   v; float    f[8]; };
static_assert(sizeof(Frag16) == 32, "frag");
static_assert(sizeof(AccF)   == 32, "acc");

__device__ __forceinline__ uint32_t pack2bf(float a, float b) {
  __bf16 x = (__bf16)a, y = (__bf16)b;
  uint16_t ux = __builtin_bit_cast(uint16_t, x);
  uint16_t uy = __builtin_bit_cast(uint16_t, y);
  return (uint32_t)ux | ((uint32_t)uy << 16);
}

__device__ __forceinline__ float fast_sigmoid(float x) {
  // 1/(1+e^-x) via v_exp_f32 (2^x) + v_rcp_f32
  float e = __builtin_amdgcn_exp2f(-1.44269504f * x);
  return __builtin_amdgcn_rcpf(1.0f + e);
}
__device__ __forceinline__ float fast_tanh(float x) {
  // tanh(x) = 1 - 2/(e^{2x}+1)
  float e = __builtin_amdgcn_exp2f(2.88539008f * x);
  return 1.0f - 2.0f * __builtin_amdgcn_rcpf(e + 1.0f);
}

__device__ __forceinline__ v8f wmma_bf16(const Frag16& a, const Frag16& b, v8f c) {
  // D(16x16,f32) = A(16x32,bf16) * B(32x16,bf16) + C
  return __builtin_amdgcn_wmma_f32_16x16x32_bf16(false, a.v, false, b.v, (short)0, c,
                                                 false, false);
}

// ---------------------------------------------------------------------------
// Layer 0: bidirectional LSTM, input dim 3. One block = 16 batch rows x 1 dir.
// Emits y0 to workspace in WMMA A-fragment bf16 layout:
//   y0frag[((bt*T + t)*4 + chunk)*256 + lane*8 + reg]  (chunk: fwd 0-1, bwd 2-3)
// ---------------------------------------------------------------------------
__global__ __launch_bounds__(256) void lstm_layer0(
    const float* __restrict__ x,
    const float* __restrict__ Wih_f, const float* __restrict__ Whh_f,
    const float* __restrict__ bih_f, const float* __restrict__ bhh_f,
    const float* __restrict__ Wih_b, const float* __restrict__ Whh_b,
    const float* __restrict__ bih_b, const float* __restrict__ bhh_b,
    uint32_t* __restrict__ y0frag)
{
  const int tid  = threadIdx.x;
  const int lane = tid & 31;
  const int wv   = tid >> 5;           // wave 0..7 -> N range [32*wv, 32*wv+32)
  const int dir  = blockIdx.x & 1;
  const int bt   = blockIdx.x >> 1;
  const int b0   = bt * TILE;

  const float* Wih = dir ? Wih_b : Wih_f;
  const float* Whh = dir ? Whh_b : Whh_f;
  const float* bih = dir ? bih_b : bih_f;
  const float* bhh = dir ? bhh_b : bhh_f;

  __shared__ float    sWih[256][4];      // Wih rows (din=3), padded
  __shared__ float    sBias[256];        // bih + bhh
  __shared__ float    sX[TILE][4];       // x_t slice
  __shared__ float    sG[TILE][GSTRIDE]; // gate pre-activations
  __shared__ uint32_t sH[2][32][8];      // h (bf16 pairs) in A-fragment layout, 2 K-chunks

  sBias[tid]   = bih[tid] + bhh[tid];
  sWih[tid][0] = Wih[tid*3 + 0];
  sWih[tid][1] = Wih[tid*3 + 1];
  sWih[tid][2] = Wih[tid*3 + 2];
  { uint32_t* hz = &sH[0][0][0]; hz[tid] = 0u; hz[tid + 256] = 0u; }

  // Whh^T B-fragments (bf16), register-resident for the whole time loop.
  // B(32x16): lane half h, reg r<4 -> K=8h+2r,+1; reg r>=4 -> K=16+8h+2(r-4),+1.
  const int nlo   = lane & 15;
  const int khalf = (lane >> 4) & 1;
  Frag16 fB[2][2];
  for (int tt = 0; tt < 2; ++tt) {
    int n = wv*32 + tt*16 + nlo;
    for (int ch = 0; ch < 2; ++ch)
      for (int r = 0; r < 8; ++r) {
        int k = ch*32 + ((r >= 4) ? 16 : 0) + khalf*8 + 2*(r & 3);
        fB[tt][ch].u[r] = pack2bf(Whh[n*H + k], Whh[n*H + k + 1]);
      }
  }

  // activation-phase fixed mapping: thread -> (row am, cols anb..anb+3) of 16x64 h/c
  const int am    = tid >> 4;
  const int anb   = (tid & 15) * 4;
  const int ach   = anb >> 5;                       // K-chunk of h
  const int akk   = anb & 31;
  const int alane = am + ((akk >> 3) & 1) * 16;     // A-fragment lane
  const int ar0   = ((akk >= 16) ? 4 : 0) + ((akk & 7) >> 1);
  float cst[4] = {0.f, 0.f, 0.f, 0.f};

  for (int s = 0; s < T; ++s) {
    const int t = dir ? (T - 1 - s) : s;
    if (tid < 48) {                                  // x is (B, 3, T)
      int m = tid / 3, c = tid % 3;
      sX[m][c] = x[(size_t)(b0 + m) * (3*T) + (size_t)c * T + t];
    }
    __syncthreads();   // sX ready; sH from previous step ready

    Frag16 fA0, fA1;
    #pragma unroll
    for (int r = 0; r < 8; ++r) { fA0.u[r] = sH[0][lane][r]; fA1.u[r] = sH[1][lane][r]; }

    const int mo = (lane >= 16) ? 8 : 0;
    #pragma unroll
    for (int tt = 0; tt < 2; ++tt) {
      int n = wv*32 + tt*16 + nlo;
      float w0 = sWih[n][0], w1 = sWih[n][1], w2 = sWih[n][2], bb = sBias[n];
      AccF a;
      #pragma unroll
      for (int r = 0; r < 8; ++r) {
        int m = r + mo;
        a.f[r] = bb + w0*sX[m][0] + w1*sX[m][1] + w2*sX[m][2];
      }
      a.v = wmma_bf16(fA0, fB[tt][0], a.v);
      a.v = wmma_bf16(fA1, fB[tt][1], a.v);
      #pragma unroll
      for (int r = 0; r < 8; ++r) sG[r + mo][n] = a.f[r];
    }
    __syncthreads();   // sG ready

    float hv[4];
    #pragma unroll
    for (int j = 0; j < 4; ++j) {
      int n = anb + j;
      float gi = fast_sigmoid(sG[am][n]);
      float gf = fast_sigmoid(sG[am][64 + n]);
      float gg = fast_tanh   (sG[am][128 + n]);
      float go = fast_sigmoid(sG[am][192 + n]);
      cst[j] = gf * cst[j] + gi * gg;
      hv[j]  = go * fast_tanh(cst[j]);
    }
    uint32_t d0 = pack2bf(hv[0], hv[1]);
    uint32_t d1 = pack2bf(hv[2], hv[3]);
    sH[ach][alane][ar0]     = d0;
    sH[ach][alane][ar0 + 1] = d1;
    // stream y0 (fwd -> chunks 0-1, bwd -> chunks 2-3) in A-fragment layout
    size_t gb = (((size_t)bt * T + (size_t)t) * 4 + (size_t)(dir*2 + ach)) * 256
              + (size_t)alane * 8 + (size_t)ar0;
    y0frag[gb]     = d0;
    y0frag[gb + 1] = d1;
  }
}

// ---------------------------------------------------------------------------
// Layer 1: bidirectional LSTM, input dim 128 (y0). Input GEMM + recurrent GEMM
// both via WMMA (K=128 -> 4 chunks, K=64 -> 2 chunks). Writes final hT only.
// ---------------------------------------------------------------------------
__global__ __launch_bounds__(256) void lstm_layer1(
    const uint32_t* __restrict__ y0frag,
    const float* __restrict__ Wih_f, const float* __restrict__ Whh_f,
    const float* __restrict__ bih_f, const float* __restrict__ bhh_f,
    const float* __restrict__ Wih_b, const float* __restrict__ Whh_b,
    const float* __restrict__ bih_b, const float* __restrict__ bhh_b,
    float* __restrict__ hcat)
{
  const int tid  = threadIdx.x;
  const int lane = tid & 31;
  const int wv   = tid >> 5;
  const int dir  = blockIdx.x & 1;
  const int bt   = blockIdx.x >> 1;
  const int b0   = bt * TILE;

  const float* Wih = dir ? Wih_b : Wih_f;
  const float* Whh = dir ? Whh_b : Whh_f;
  const float* bih = dir ? bih_b : bih_f;
  const float* bhh = dir ? bhh_b : bhh_f;

  __shared__ float    sBias[256];
  __shared__ float    sG[TILE][GSTRIDE];
  __shared__ uint32_t sH[2][32][8];

  sBias[tid] = bih[tid] + bhh[tid];
  { uint32_t* hz = &sH[0][0][0]; hz[tid] = 0u; hz[tid + 256] = 0u; }

  const int nlo   = lane & 15;
  const int khalf = (lane >> 4) & 1;
  Frag16 fBi[2][4], fBh[2][2];
  for (int tt = 0; tt < 2; ++tt) {
    int n = wv*32 + tt*16 + nlo;
    for (int ch = 0; ch < 4; ++ch)
      for (int r = 0; r < 8; ++r) {
        int k = ch*32 + ((r >= 4) ? 16 : 0) + khalf*8 + 2*(r & 3);
        fBi[tt][ch].u[r] = pack2bf(Wih[n*128 + k], Wih[n*128 + k + 1]);
      }
    for (int ch = 0; ch < 2; ++ch)
      for (int r = 0; r < 8; ++r) {
        int k = ch*32 + ((r >= 4) ? 16 : 0) + khalf*8 + 2*(r & 3);
        fBh[tt][ch].u[r] = pack2bf(Whh[n*H + k], Whh[n*H + k + 1]);
      }
  }

  const int am    = tid >> 4;
  const int anb   = (tid & 15) * 4;
  const int ach   = anb >> 5;
  const int akk   = anb & 31;
  const int alane = am + ((akk >> 3) & 1) * 16;
  const int ar0   = ((akk >= 16) ? 4 : 0) + ((akk & 7) >> 1);
  float cst[4] = {0.f, 0.f, 0.f, 0.f};
  float hv[4]  = {0.f, 0.f, 0.f, 0.f};

  for (int s = 0; s < T; ++s) {
    const int t = dir ? (T - 1 - s) : s;
    // y0_t A-fragments: contiguous 32B per chunk per lane -> global_load_b128
    const uint32_t* yb = y0frag + ((size_t)(bt * T + t) * 4) * 256 + (size_t)lane * 8;
    Frag16 fX[4];
    #pragma unroll
    for (int ch = 0; ch < 4; ++ch)
      #pragma unroll
      for (int r = 0; r < 8; ++r) fX[ch].u[r] = yb[(size_t)ch * 256 + r];

    __syncthreads();
    Frag16 fA0, fA1;
    #pragma unroll
    for (int r = 0; r < 8; ++r) { fA0.u[r] = sH[0][lane][r]; fA1.u[r] = sH[1][lane][r]; }

    const int mo = (lane >= 16) ? 8 : 0;
    #pragma unroll
    for (int tt = 0; tt < 2; ++tt) {
      int n = wv*32 + tt*16 + nlo;
      float bb = sBias[n];
      AccF a;
      #pragma unroll
      for (int r = 0; r < 8; ++r) a.f[r] = bb;
      a.v = wmma_bf16(fX[0], fBi[tt][0], a.v);
      a.v = wmma_bf16(fX[1], fBi[tt][1], a.v);
      a.v = wmma_bf16(fX[2], fBi[tt][2], a.v);
      a.v = wmma_bf16(fX[3], fBi[tt][3], a.v);
      a.v = wmma_bf16(fA0,  fBh[tt][0], a.v);
      a.v = wmma_bf16(fA1,  fBh[tt][1], a.v);
      #pragma unroll
      for (int r = 0; r < 8; ++r) sG[r + mo][n] = a.f[r];
    }
    __syncthreads();

    #pragma unroll
    for (int j = 0; j < 4; ++j) {
      int n = anb + j;
      float gi = fast_sigmoid(sG[am][n]);
      float gf = fast_sigmoid(sG[am][64 + n]);
      float gg = fast_tanh   (sG[am][128 + n]);
      float go = fast_sigmoid(sG[am][192 + n]);
      cst[j] = gf * cst[j] + gi * gg;
      hv[j]  = go * fast_tanh(cst[j]);
    }
    sH[ach][alane][ar0]     = pack2bf(hv[0], hv[1]);
    sH[ach][alane][ar0 + 1] = pack2bf(hv[2], hv[3]);
  }
  // final hidden state hT -> hcat (B x 128), fwd cols 0-63, bwd cols 64-127
  #pragma unroll
  for (int j = 0; j < 4; ++j)
    hcat[(size_t)(b0 + am) * 128 + (size_t)dir * 64 + anb + j] = hv[j];
}

// ---------------------------------------------------------------------------
// FC head: out = relu(hcat @ fc1_W^T + b1) @ fc2_W^T + b2   (B x 2)
// ---------------------------------------------------------------------------
__global__ __launch_bounds__(64) void fc_head(
    const float* __restrict__ hcat,
    const float* __restrict__ fc1_W, const float* __restrict__ fc1_b,
    const float* __restrict__ fc2_W, const float* __restrict__ fc2_b,
    float* __restrict__ out)
{
  __shared__ float row[128];
  __shared__ float h1[64];
  const int rix = blockIdx.x;
  const int tid = threadIdx.x;
  row[tid]      = hcat[(size_t)rix * 128 + tid];
  row[tid + 64] = hcat[(size_t)rix * 128 + tid + 64];
  __syncthreads();
  float a = fc1_b[tid];
  #pragma unroll 8
  for (int k = 0; k < 128; ++k) a += row[k] * fc1_W[tid * 128 + k];
  h1[tid] = fmaxf(a, 0.0f);
  __syncthreads();
  if (tid < 2) {
    float a2 = fc2_b[tid];
    for (int k = 0; k < 64; ++k) a2 += h1[k] * fc2_W[tid * 64 + k];
    out[(size_t)rix * 2 + tid] = a2;
  }
}

extern "C" void kernel_launch(void* const* d_in, const int* in_sizes, int n_in,
                              void* d_out, int out_size, void* d_ws, size_t ws_size,
                              hipStream_t stream) {
  (void)in_sizes; (void)n_in; (void)out_size; (void)ws_size;
  const float* x = (const float*)d_in[0];

  // workspace layout: y0 fragments (bf16 pairs) then hcat (f32)
  uint32_t* y0frag = (uint32_t*)d_ws;
  const size_t y0_dwords = (size_t)NTILES * T * 4 * 256;       // 33,554,432 dwords = 128 MiB
  float* hcat = (float*)((char*)d_ws + y0_dwords * sizeof(uint32_t));

  lstm_layer0<<<NTILES * 2, 256, 0, stream>>>(
      x,
      (const float*)d_in[1], (const float*)d_in[2], (const float*)d_in[3], (const float*)d_in[4],
      (const float*)d_in[5], (const float*)d_in[6], (const float*)d_in[7], (const float*)d_in[8],
      y0frag);

  lstm_layer1<<<NTILES * 2, 256, 0, stream>>>(
      y0frag,
      (const float*)d_in[9],  (const float*)d_in[10], (const float*)d_in[11], (const float*)d_in[12],
      (const float*)d_in[13], (const float*)d_in[14], (const float*)d_in[15], (const float*)d_in[16],
      hcat);

  fc_head<<<BATCH, 64, 0, stream>>>(
      hcat,
      (const float*)d_in[17], (const float*)d_in[18],
      (const float*)d_in[19], (const float*)d_in[20],
      (float*)d_out);
}
